// SwinTransformerBlock_7567732375800
// MI455X (gfx1250) — matrix operand order repaired and checked by reading, not compiled, not run on hardware
//
#include <hip/hip_runtime.h>
#include <hip/hip_bf16.h>
#include <math.h>

// ---------------- problem constants (from reference) ----------------
#define T_TOK  401408      // 8 * 224 * 224 tokens
#define C_DIM  192
#define HID    768
#define NHEAD  6
#define NWIN   25088       // 8 * 56 * 56 windows of 4x4
#define IMG    224
#define SHIFT2 2

// ---------------- CDNA5 WMMA types ----------------
typedef __attribute__((ext_vector_type(16))) __bf16 v16bf;
typedef __attribute__((ext_vector_type(8)))  float  v8f;

union AFrag { v16bf v; unsigned u[8]; };

__device__ __forceinline__ unsigned short f2bf(float f) {
  unsigned u = __float_as_uint(f);
  u += 0x7FFFu + ((u >> 16) & 1u);          // round-to-nearest-even
  return (unsigned short)(u >> 16);
}

// ---------------- CDNA5 async-to-LDS helpers (ASYNCcnt-tracked DMA path) ----------------
__device__ __forceinline__ void async_b128(unsigned lds_off, const void* gaddr) {
  asm volatile("global_load_async_to_lds_b128 %0, %1, off"
               :: "v"(lds_off), "v"(gaddr) : "memory");
}
__device__ __forceinline__ void wait_async0() {
  asm volatile("s_wait_asynccnt 0x0" ::: "memory");
}

// ---------------- index kernel: shift + window partition bijection ----------------
__global__ void k_build_idx(int* __restrict__ idx) {
  int r = blockIdx.x * 256 + threadIdx.x;
  if (r >= T_TOK) return;
  int w = r >> 4, t = r & 15;
  int b = w / 3136, rem = w % 3136;
  int wy = rem / 56, wx = rem % 56;
  int ty = t >> 2, tx = t & 3;
  int sh = (wy * 4 + ty + SHIFT2) % IMG;
  int sw = (wx * 4 + tx + SHIFT2) % IMG;
  idx[r] = b * (IMG * IMG) + sh * IMG + sw;
}

// ---------------- fp32 -> bf16 weight conversion ----------------
__global__ void k_cvt_bf16(const float* __restrict__ src, unsigned short* __restrict__ dst, int n) {
  int i = blockIdx.x * 256 + threadIdx.x;
  if (i < n) dst[i] = f2bf(src[i]);
}

// ---------------- LayerNorm: one wave per row (C=192 = 6 per lane) ----------------
__global__ void k_layernorm(const float* __restrict__ x, const float* __restrict__ w,
                            const float* __restrict__ b, unsigned short* __restrict__ out) {
  int wid = threadIdx.x >> 5, lane = threadIdx.x & 31;
  long long row = (long long)blockIdx.x * 8 + wid;
  if (row >= T_TOK) return;
  const float* xr = x + row * C_DIM;
  float v[6]; float s = 0.f;
  #pragma unroll
  for (int j = 0; j < 6; ++j) { v[j] = xr[lane + 32 * j]; s += v[j]; }
  #pragma unroll
  for (int m = 16; m > 0; m >>= 1) s += __shfl_xor(s, m, 32);
  float mean = s * (1.0f / C_DIM);
  float q = 0.f;
  #pragma unroll
  for (int j = 0; j < 6; ++j) { float d = v[j] - mean; q += d * d; }
  #pragma unroll
  for (int m = 16; m > 0; m >>= 1) q += __shfl_xor(q, m, 32);
  float rstd = rsqrtf(q * (1.0f / C_DIM) + 1e-5f);
  unsigned short* orow = out + row * C_DIM;
  #pragma unroll
  for (int j = 0; j < 6; ++j) {
    int c = lane + 32 * j;
    orow[c] = f2bf((v[j] - mean) * rstd * w[c] + b[c]);
  }
}

// ---------------- tiled WMMA GEMM: out[M,N] = A[M,K] @ B[N,K]^T + bias ----------------
// block = 256 thr (8 waves). Block tile 128(M) x 64(N); each wave computes a 32x32
// macro-tile = 2 A-frags x 2 B-frags -> 4 WMMAs per 32-deep k-step.
// Tiles staged via global_load_async_to_lds_b128 with double-buffered LDS.
#define EPI_BF16 0
#define EPI_GELU 1
#define EPI_RES  2

template<int EPI>
__global__ void k_gemm(const unsigned short* __restrict__ A,
                       const unsigned short* __restrict__ Bw, int K,
                       const float* __restrict__ bias,
                       unsigned short* __restrict__ outBf,
                       const float* __restrict__ residual,
                       float* __restrict__ outF,
                       const int* __restrict__ aIdx,
                       const int* __restrict__ oIdx,
                       int N) {
  // stride 40 halves (80 B): 16B-aligned rows, staggered across banks
  __shared__ __align__(16) unsigned short As[2][128 * 40];
  __shared__ __align__(16) unsigned short Bs[2][64 * 40];
  const int tid = threadIdx.x;
  const int waveId = tid >> 5, lane = tid & 31;
  const int mSub = waveId & 3, nSub = waveId >> 2;   // 4 x 2 wave grid of 32x32 tiles
  const int hi = lane >> 4, lil = lane & 15;
  const long long blockM = (long long)blockIdx.y * 128;
  const int nBase = blockIdx.x * 64;

  // ---- loader assignments (16B = 8 bf16 chunks) ----
  // A tile: 128 rows x 32 cols = 512 chunks -> 2 per thread
  const int ar0 = tid >> 2;              // rows 0..63
  const int ar1 = 64 + (tid >> 2);       // rows 64..127
  const int ach = (tid & 3) * 8;
  long long arow0 = blockM + ar0, arow1 = blockM + ar1;
  if (aIdx) { arow0 = aIdx[arow0]; arow1 = aIdx[arow1]; }
  const unsigned short* aSrc0 = A + arow0 * (long long)K;
  const unsigned short* aSrc1 = A + arow1 * (long long)K;
  // B tile: 64 rows x 32 cols = 256 chunks -> 1 per thread
  const int br = tid >> 2, bch = (tid & 3) * 8;
  const unsigned short* bSrc = Bw + (long long)(nBase + br) * K;

  const unsigned ldsA0 = (unsigned)(uintptr_t)&As[0][0];
  const unsigned ldsB0 = (unsigned)(uintptr_t)&Bs[0][0];
  const unsigned aslotOff0 = (unsigned)(ar0 * 40 + ach) * 2u;
  const unsigned aslotOff1 = (unsigned)(ar1 * 40 + ach) * 2u;
  const unsigned bslotOff  = (unsigned)(br * 40 + bch) * 2u;

  auto issueTile = [&](int buf, int k0) {
    unsigned aBase = ldsA0 + (unsigned)buf * (128u * 40u * 2u);
    unsigned bBase = ldsB0 + (unsigned)buf * (64u * 40u * 2u);
    async_b128(aBase + aslotOff0, aSrc0 + k0 + ach);
    async_b128(aBase + aslotOff1, aSrc1 + k0 + ach);
    async_b128(bBase + bslotOff,  bSrc  + k0 + bch);
  };

  v8f acc[2][2] = {};
  const int nk = K / 32;
  issueTile(0, 0);
  for (int t = 0; t < nk; ++t) {
    const int cur = t & 1;
    wait_async0();          // this wave's staged tile (and any prefetch) is in LDS
    __syncthreads();        // everyone finished reading the buffer we are about to reuse
    if (t + 1 < nk) issueTile(cur ^ 1, (t + 1) * 32);

    const unsigned short* as = &As[cur][0];
    const unsigned short* bs = &Bs[cur][0];
    AFrag a[2], b[2];
    #pragma unroll
    for (int i = 0; i < 2; ++i) {
      const int mrow = mSub * 32 + i * 16 + lil;   // A layout: M = lane&15 in both halves
      #pragma unroll
      for (int vv = 0; vv < 8; ++vv) {   // A layout: K = hi*16 + 2v{+1} (contiguous, matches B)
        int kk = hi * 16 + vv * 2;
        a[i].u[vv] = *(const unsigned*)(as + mrow * 40 + kk);
      }
    }
    #pragma unroll
    for (int j = 0; j < 2; ++j) {
      const int ncol = nSub * 32 + j * 16 + lil;   // B layout: lanes 0-15 K=0..15, 16-31 K=16..31
      #pragma unroll
      for (int vv = 0; vv < 8; ++vv) {
        int kk = hi * 16 + vv * 2;
        b[j].u[vv] = *(const unsigned*)(bs + ncol * 40 + kk);
      }
    }
    #pragma unroll
    for (int i = 0; i < 2; ++i)
      #pragma unroll
      for (int j = 0; j < 2; ++j)
        acc[i][j] = __builtin_amdgcn_wmma_f32_16x16x32_bf16(false, a[i].v, false, b[j].v,
                                                            (short)0, acc[i][j], false, false);
  }
  // epilogue: D element vv -> row (vv + 8*hi), col lil
  #pragma unroll
  for (int i = 0; i < 2; ++i) {
    #pragma unroll
    for (int j = 0; j < 2; ++j) {
      const int gn = nBase + nSub * 32 + j * 16 + lil;
      const float bn = bias[gn];
      #pragma unroll
      for (int vv = 0; vv < 8; ++vv) {
        long long gm = blockM + mSub * 32 + i * 16 + vv + 8 * hi;
        float val = acc[i][j][vv] + bn;
        if (EPI == EPI_GELU) {
          val = 0.5f * val * (1.0f + erff(val * 0.70710678118654752f));
          outBf[gm * N + gn] = f2bf(val);
        } else if (EPI == EPI_BF16) {
          outBf[gm * N + gn] = f2bf(val);
        } else {
          long long orow = oIdx ? (long long)oIdx[gm] : gm;
          outF[orow * N + gn] = val + residual[orow * N + gn];
        }
      }
    }
  }
}

// ---------------- windowed attention: one wave per (window, head) ----------------
// qkv row layout per window token: [3][NHEAD][32] bf16 (576 ch). S=QK^T is one WMMA,
// softmax in D-layout registers, P transposed via LDS into A-layout (K padded to 32),
// P@V is two WMMAs (two 16-wide halves of hd=32).
__global__ void k_attn(const unsigned short* __restrict__ qkv,
                       const float* __restrict__ rpb,
                       unsigned short* __restrict__ out) {
  __shared__ float P[NHEAD * 256];
  const int head = threadIdx.x >> 5, lane = threadIdx.x & 31;
  const int hi = lane >> 4, lil = lane & 15;
  const long long row0 = (long long)blockIdx.x * 16;
  const unsigned short* base = qkv + row0 * 576;
  const int qoff = head * 32, koff = 192 + head * 32, voff = 384 + head * 32;

  AFrag aQ, bK;
  #pragma unroll
  for (int vv = 0; vv < 8; ++vv) {
    int kk = hi * 16 + vv * 2;      // d index, even
    aQ.u[vv] = *(const unsigned*)(base + lil * 576 + qoff + kk);
  }
  #pragma unroll
  for (int vv = 0; vv < 8; ++vv) {
    int dd = hi * 16 + vv * 2;                                  // K^T: lane reads row m=lil of K
    bK.u[vv] = *(const unsigned*)(base + lil * 576 + koff + dd);
  }
  v8f s = {};
  s = __builtin_amdgcn_wmma_f32_16x16x32_bf16(false, aQ.v, false, bK.v, (short)0, s, false, false);

  const float scale = 0.17677669529663687f;   // 32^-0.5
  float p[8];
  #pragma unroll
  for (int vv = 0; vv < 8; ++vv) {
    int n = vv + 8 * hi, m = lil;
    int rel = ((n >> 2) - (m >> 2) + 3) * 7 + ((n & 3) - (m & 3) + 3);
    p[vv] = s[vv] * scale + rpb[rel * NHEAD + head];
  }
  // row softmax: columns live across the 16 lanes of this half (xor masks 8..1 stay in-half)
  #pragma unroll
  for (int vv = 0; vv < 8; ++vv) {
    float mx = p[vv];
    #pragma unroll
    for (int msk = 8; msk > 0; msk >>= 1) mx = fmaxf(mx, __shfl_xor(mx, msk, 32));
    float e = __expf(p[vv] - mx);
    float sm = e;
    #pragma unroll
    for (int msk = 8; msk > 0; msk >>= 1) sm += __shfl_xor(sm, msk, 32);
    p[vv] = e / sm;
  }
  float* Ph = &P[head * 256];
  #pragma unroll
  for (int vv = 0; vv < 8; ++vv) Ph[(vv + 8 * hi) * 16 + lil] = p[vv];
  __syncthreads();
  // P as A-matrix (16x32, K=16..31 zero-padded)
  AFrag aP;
  #pragma unroll
  for (int vv = 0; vv < 8; ++vv) {
    int kk = hi * 16 + vv * 2;
    unsigned lo = (kk < 16) ? (unsigned)f2bf(Ph[lil * 16 + kk])     : 0u;
    unsigned hh = (kk + 1 < 16) ? (unsigned)f2bf(Ph[lil * 16 + kk + 1]) : 0u;
    aP.u[vv] = lo | (hh << 16);
  }
  // V as B-matrix for the two 16-wide d halves; K rows 16..31 (lanes 16-31) are zero
  AFrag bV0, bV1;
  #pragma unroll
  for (int vv = 0; vv < 8; ++vv) {
    int m0 = hi * 16 + vv * 2;
    unsigned w0 = 0, w1 = 0;
    if (hi == 0) {
      unsigned a0 = base[(m0    ) * 576 + voff +      lil];
      unsigned a1 = base[(m0 + 1) * 576 + voff +      lil];
      unsigned c0 = base[(m0    ) * 576 + voff + 16 + lil];
      unsigned c1 = base[(m0 + 1) * 576 + voff + 16 + lil];
      w0 = a0 | (a1 << 16);
      w1 = c0 | (c1 << 16);
    }
    bV0.u[vv] = w0; bV1.u[vv] = w1;
  }
  v8f o0 = {}, o1 = {};
  o0 = __builtin_amdgcn_wmma_f32_16x16x32_bf16(false, aP.v, false, bV0.v, (short)0, o0, false, false);
  o1 = __builtin_amdgcn_wmma_f32_16x16x32_bf16(false, aP.v, false, bV1.v, (short)0, o1, false, false);
  #pragma unroll
  for (int vv = 0; vv < 8; ++vv) {
    long long n = vv + 8 * hi;
    out[(row0 + n) * C_DIM + head * 32 +      lil] = f2bf(o0[vv]);
    out[(row0 + n) * C_DIM + head * 32 + 16 + lil] = f2bf(o1[vv]);
  }
}

// ---------------- launch ----------------
extern "C" void kernel_launch(void* const* d_in, const int* in_sizes, int n_in,
                              void* d_out, int out_size, void* d_ws, size_t ws_size,
                              hipStream_t stream) {
  const float* x     = (const float*)d_in[0];
  const float* n1w   = (const float*)d_in[1];
  const float* n1b   = (const float*)d_in[2];
  const float* qkvw  = (const float*)d_in[3];
  const float* qkvb  = (const float*)d_in[4];
  const float* rpb   = (const float*)d_in[5];
  const float* projw = (const float*)d_in[6];
  const float* projb = (const float*)d_in[7];
  const float* n2w   = (const float*)d_in[8];
  const float* n2b   = (const float*)d_in[9];
  const float* fc1w  = (const float*)d_in[10];
  const float* fc1b  = (const float*)d_in[11];
  const float* fc2w  = (const float*)d_in[12];
  const float* fc2b  = (const float*)d_in[13];
  float* out = (float*)d_out;
  char* ws = (char*)d_ws;

  // workspace layout (~1.01 GiB, all 16B-aligned)
  const size_t IDX_OFF = 0;
  const size_t WQ_OFF  = IDX_OFF + (size_t)T_TOK * 4;                 // gather idx
  const size_t WP_OFF  = WQ_OFF + (size_t)576 * 192 * 2;              // qkv_w bf16
  const size_t W1_OFF  = WP_OFF + (size_t)192 * 192 * 2;              // proj_w bf16
  const size_t W2_OFF  = W1_OFF + (size_t)768 * 192 * 2;              // fc1_w bf16
  const size_t A_OFF   = W2_OFF + (size_t)192 * 768 * 2;              // fc2_w bf16 end
  const size_t X1_OFF  = A_OFF + (size_t)T_TOK * C_DIM * 2;           // A: hn/attnout/ln2 (bf16)
  const size_t BIG_OFF = X1_OFF + (size_t)T_TOK * C_DIM * 4;          // x1 (f32)

  int* idx = (int*)(ws + IDX_OFF);
  unsigned short* wqkv = (unsigned short*)(ws + WQ_OFF);
  unsigned short* wproj = (unsigned short*)(ws + WP_OFF);
  unsigned short* wfc1 = (unsigned short*)(ws + W1_OFF);
  unsigned short* wfc2 = (unsigned short*)(ws + W2_OFF);
  unsigned short* bufA = (unsigned short*)(ws + A_OFF);     // hn -> attnout -> ln2out
  float* x1 = (float*)(ws + X1_OFF);
  unsigned short* bufBig = (unsigned short*)(ws + BIG_OFF); // qkv -> mlp hidden

  k_build_idx<<<(T_TOK + 255) / 256, 256, 0, stream>>>(idx);
  k_cvt_bf16<<<(576 * 192 + 255) / 256, 256, 0, stream>>>(qkvw, wqkv, 576 * 192);
  k_cvt_bf16<<<(192 * 192 + 255) / 256, 256, 0, stream>>>(projw, wproj, 192 * 192);
  k_cvt_bf16<<<(768 * 192 + 255) / 256, 256, 0, stream>>>(fc1w, wfc1, 768 * 192);
  k_cvt_bf16<<<(192 * 768 + 255) / 256, 256, 0, stream>>>(fc2w, wfc2, 192 * 768);

  // 1. LN1 -> bf16
  k_layernorm<<<T_TOK / 8, 256, 0, stream>>>(x, n1w, n1b, bufA);
  // 2. QKV with window gather (rows in window order)
  k_gemm<EPI_BF16><<<dim3(576 / 64, T_TOK / 128), 256, 0, stream>>>(
      bufA, wqkv, 192, qkvb, bufBig, nullptr, nullptr, idx, nullptr, 576);
  // 3. attention per (window, head) -> attnout (reuses bufA)
  k_attn<<<NWIN, NHEAD * 32, 0, stream>>>(bufBig, rpb, bufA);
  // 4. proj + window-reverse scatter + residual -> x1 (f32)
  k_gemm<EPI_RES><<<dim3(192 / 64, T_TOK / 128), 256, 0, stream>>>(
      bufA, wproj, 192, projb, nullptr, x, x1, nullptr, idx, 192);
  // 5. LN2 -> bf16 (reuses bufA)
  k_layernorm<<<T_TOK / 8, 256, 0, stream>>>(x1, n2w, n2b, bufA);
  // 6. fc1 + exact GELU -> hidden (reuses bufBig)
  k_gemm<EPI_GELU><<<dim3(768 / 64, T_TOK / 128), 256, 0, stream>>>(
      bufA, wfc1, 192, fc1b, bufBig, nullptr, nullptr, nullptr, nullptr, 768);
  // 7. fc2 + residual(x1) -> d_out (f32)
  k_gemm<EPI_RES><<<dim3(192 / 64, T_TOK / 128), 256, 0, stream>>>(
      bufBig, wfc2, 768, fc2b, nullptr, x1, out, nullptr, nullptr, 192);
}
